// MultiLatentAttention_77764677861496
// MI455X (gfx1250) — matrix operand order repaired and checked by reading, not compiled, not run on hardware
//
#include <hip/hip_runtime.h>
#include <hip/hip_bf16.h>
#include <cstdint>
#include <cstddef>

// ---- problem dims (fixed by reference) ----
#define N_EMBD 1024
#define LATENT 512
#define HEADS  16
#define HDIM   64
#define BATCH  4
#define TLEN   2048
#define BTOT   (BATCH*TLEN)          // 8192 rows
#define NUP    (N_EMBD + 2*HDIM)     // 1152 = q|k|v columns

typedef __attribute__((ext_vector_type(16))) __bf16 v16bf;
typedef __attribute__((ext_vector_type(8)))  __bf16 v8bf;
typedef __attribute__((ext_vector_type(4)))  __bf16 v4bf;
typedef __attribute__((ext_vector_type(8)))  float  v8f;

static __device__ __forceinline__ v16bf cat8(v8bf lo, v8bf hi) {
  return __builtin_shufflevector(lo, hi, 0,1,2,3,4,5,6,7,8,9,10,11,12,13,14,15);
}
static __device__ __forceinline__ v8f wmma_bf16(v16bf a, v16bf b, v8f c) {
  return __builtin_amdgcn_wmma_f32_16x16x32_bf16(false, a, false, b, (short)0, c,
                                                 false, false);
}

// ============================================================
// prep: fp32 -> bf16 (x, Wc_down, Wc_proj) + concat Wq|Wkv -> [512][1152]
// ============================================================
__global__ __launch_bounds__(256) void prep_kernel(
    const float* __restrict__ x, const float* __restrict__ Wd,
    const float* __restrict__ Wq, const float* __restrict__ Wkv,
    const float* __restrict__ Wp,
    __bf16* __restrict__ xbf, __bf16* __restrict__ wdbf,
    __bf16* __restrict__ wupbf, __bf16* __restrict__ wpbf)
{
  size_t i = (size_t)blockIdx.x * blockDim.x + threadIdx.x;
  if (i < (size_t)BTOT * N_EMBD)   xbf[i] = (__bf16)x[i];
  if (i < (size_t)N_EMBD * LATENT) wdbf[i] = (__bf16)Wd[i];
  if (i < (size_t)N_EMBD * N_EMBD) wpbf[i] = (__bf16)Wp[i];
  if (i < (size_t)LATENT * NUP) {
    int k = (int)(i / NUP), c = (int)(i % NUP);
    float v = (c < N_EMBD) ? Wq[(size_t)k * N_EMBD + c]
                           : Wkv[(size_t)k * (2*HDIM) + (c - N_EMBD)];
    wupbf[i] = (__bf16)v;
  }
}

// ============================================================
// WMMA GEMM: C(MxN) = A(MxK) @ W(KxN), bf16 in, f32 acc.
// 256 threads (8 waves, 32 rows each) -> tile 256(M) x 64(N), K step 32.
// Software-pipelined: next W chunk + next A fragments prefetched into
// registers while current chunk's 8 WMMAs execute.
// mode 0: bf16 row-major out (Cb, stride N)
// mode 1: f32  row-major out (Cf, stride N)
// mode 2: N==1152 split: [0,1024)->Cb(q), [1024,1088)->Kb, [1088,1152)->VTb[b][d][t]
// ============================================================
__global__ __launch_bounds__(256) void wmma_gemm_kernel(
    const __bf16* __restrict__ A, const __bf16* __restrict__ W,
    float* __restrict__ Cf, __bf16* __restrict__ Cb,
    __bf16* __restrict__ Kb, __bf16* __restrict__ VTb,
    int M, int N, int K, int mode)
{
  __shared__ __bf16 wt[64][40];   // transposed W chunk [n_local][k_local], +8 pad
  const int tid = threadIdx.x;
  const int wv  = tid >> 5, lane = tid & 31;
  const int n16 = lane & 15, hlf = lane >> 4;
  const int rowBase = blockIdx.x * 256 + wv * 32;
  const int colBase = blockIdx.y * 64;
  const int kk = tid >> 3;          // 0..31 : k row of W chunk
  const int cc = (tid & 7) * 8;     // 0..56 : col fragment of W chunk

  v8f acc[2][4];
  #pragma unroll
  for (int j = 0; j < 2; ++j)
    #pragma unroll
    for (int t = 0; t < 4; ++t)
      #pragma unroll
      for (int r = 0; r < 8; ++r) acc[j][t][r] = 0.0f;

  const __bf16* arow0 = A + (size_t)(rowBase + n16) * K;
  const __bf16* arow1 = A + (size_t)(rowBase + 16 + n16) * K;

  // ---- prologue: load chunk 0 into registers ----
  v8bf wreg = *(const v8bf*)(W + (size_t)kk * N + colBase + cc);
  v8bf a0lo = *(const v8bf*)(arow0 + hlf * 8);
  v8bf a0hi = *(const v8bf*)(arow0 + 16 + hlf * 8);
  v8bf a1lo = *(const v8bf*)(arow1 + hlf * 8);
  v8bf a1hi = *(const v8bf*)(arow1 + 16 + hlf * 8);

  for (int k0 = 0; k0 < K; k0 += 32) {
    __syncthreads();                       // prior iter's LDS reads done
    #pragma unroll
    for (int j = 0; j < 8; ++j) wt[cc + j][kk] = wreg[j];   // publish W chunk
    __syncthreads();

    // current-chunk A fragments (ISA 16-bit A-layout)
    const v16bf av0 = cat8(a0lo, a0hi);
    const v16bf av1 = cat8(a1lo, a1hi);

    // prefetch next chunk into registers (latency hidden behind WMMAs)
    if (k0 + 32 < K) {
      const int kn = k0 + 32;
      wreg = *(const v8bf*)(W + (size_t)(kn + kk) * N + colBase + cc);
      a0lo = *(const v8bf*)(arow0 + kn + hlf * 8);
      a0hi = *(const v8bf*)(arow0 + kn + 16 + hlf * 8);
      a1lo = *(const v8bf*)(arow1 + kn + hlf * 8);
      a1hi = *(const v8bf*)(arow1 + kn + 16 + hlf * 8);
    }

    #pragma unroll
    for (int nt = 0; nt < 4; ++nt) {
      // B layout: lane col n16; elem e<8 -> K = hlf*8+e, e>=8 -> K = 16+hlf*8+(e-8)
      const v16bf bv = cat8(*(const v8bf*)&wt[nt * 16 + n16][hlf * 8],
                            *(const v8bf*)&wt[nt * 16 + n16][16 + hlf * 8]);
      acc[0][nt] = wmma_bf16(av0, bv, acc[0][nt]);   // B reused x2 from regs
      acc[1][nt] = wmma_bf16(av1, bv, acc[1][nt]);
    }
  }

  #pragma unroll
  for (int j = 0; j < 2; ++j) {
    #pragma unroll
    for (int nt = 0; nt < 4; ++nt) {
      #pragma unroll
      for (int r = 0; r < 8; ++r) {
        const int row = rowBase + j * 16 + hlf * 8 + r;   // C layout: M=hlf*8+r
        const int col = colBase + nt * 16 + n16;          // N = lane%16
        const float v = acc[j][nt][r];
        if (mode == 1) {
          Cf[(size_t)row * N + col] = v;
        } else if (mode == 0) {
          Cb[(size_t)row * N + col] = (__bf16)v;
        } else {
          if (col < N_EMBD) {
            Cb[(size_t)row * N_EMBD + col] = (__bf16)v;
          } else if (col < N_EMBD + HDIM) {
            Kb[(size_t)row * HDIM + (col - N_EMBD)] = (__bf16)v;
          } else {
            const int d = col - (N_EMBD + HDIM);
            const int bb = row / TLEN, t = row % TLEN;
            VTb[((size_t)bb * HDIM + d) * TLEN + t] = (__bf16)v;  // v transposed
          }
        }
      }
    }
  }
}

// ============================================================
// Flash attention: 1 workgroup = 16 waves = 16 heads, one 16-row t-block.
// K/V (shared across heads in MLA) staged once in LDS per 32-wide s-block,
// with register double-buffering of the next s-block's fragments.
// ============================================================
__global__ __launch_bounds__(512) void mla_attn_kernel(
    const __bf16* __restrict__ Q,   // [BT][1024] bf16
    const __bf16* __restrict__ Kb,  // [BT][64]
    const __bf16* __restrict__ VT,  // [B][64][T]
    __bf16* __restrict__ Y)         // [BT][1024] bf16
{
  __shared__ __bf16 klds[32][72];          // [s_local][d], +8 pad
  __shared__ __bf16 vlds[64][40];          // [d][s_local], +8 pad
  __shared__ __bf16 plds[HEADS][16][40];   // per-wave P tile [t][s], +8 pad

  const int tid  = threadIdx.x;
  const int h    = tid >> 5, lane = tid & 31;
  const int n16  = lane & 15, hlf = lane >> 4;
  const int bidx = blockIdx.x / (TLEN / 16);
  const int tblk = blockIdx.x % (TLEN / 16);
  const int t0   = tblk * 16;
  const size_t btBase = (size_t)bidx * TLEN;

  // cooperative-load fragment coordinates (fixed per thread)
  const int rk = tid >> 4, ck = (tid & 15) * 4;   // K tile 32x64
  const int rv = tid >> 3, cv = (tid & 7) * 4;    // V^T tile 64x32

  // Q in A-layout, two K-chunks over d (0..31, 32..63)
  const __bf16* qrow = Q + (btBase + t0 + n16) * N_EMBD + h * HDIM;
  const v16bf qa0 = cat8(*(const v8bf*)(qrow + hlf * 8),
                         *(const v8bf*)(qrow + 16 + hlf * 8));
  const v16bf qa1 = cat8(*(const v8bf*)(qrow + 32 + hlf * 8),
                         *(const v8bf*)(qrow + 48 + hlf * 8));

  v8f oacc[4];
  float mrun[8], lrun[8];
  #pragma unroll
  for (int t = 0; t < 4; ++t)
    #pragma unroll
    for (int r = 0; r < 8; ++r) oacc[t][r] = 0.0f;
  #pragma unroll
  for (int r = 0; r < 8; ++r) { mrun[r] = -1e30f; lrun[r] = 0.0f; }

  const int tmax = t0 + 15;
  const int nsb  = tmax / 32 + 1;
  const float scale = 0.125f;               // 1/sqrt(64)

  // prologue: fragments of s-block 0
  v4bf kreg = *(const v4bf*)(Kb + (btBase + rk) * HDIM + ck);
  v4bf vreg = *(const v4bf*)(VT + ((size_t)bidx * HDIM + rv) * TLEN + cv);

  for (int sb = 0; sb < nsb; ++sb) {
    const int s0 = sb * 32;
    __syncthreads();
    *(v4bf*)&klds[rk][ck] = kreg;           // publish K/V tiles
    *(v4bf*)&vlds[rv][cv] = vreg;
    __syncthreads();
    if (sb + 1 < nsb) {                     // prefetch next s-block
      const int sn = s0 + 32;
      kreg = *(const v4bf*)(Kb + (btBase + sn + rk) * HDIM + ck);
      vreg = *(const v4bf*)(VT + ((size_t)bidx * HDIM + rv) * TLEN + sn + cv);
    }

    // S = Q K^T : two 16x16 tiles (s cols 0..15, 16..31), 2 wmma each (d chunks)
    v8f s[2];
    #pragma unroll
    for (int st = 0; st < 2; ++st) {
      v8f a;
      #pragma unroll
      for (int r = 0; r < 8; ++r) a[r] = 0.0f;
      v16bf b0 = cat8(*(const v8bf*)&klds[st * 16 + n16][hlf * 8],
                      *(const v8bf*)&klds[st * 16 + n16][16 + hlf * 8]);
      a = wmma_bf16(qa0, b0, a);
      v16bf b1 = cat8(*(const v8bf*)&klds[st * 16 + n16][32 + hlf * 8],
                      *(const v8bf*)&klds[st * 16 + n16][48 + hlf * 8]);
      a = wmma_bf16(qa1, b1, a);
      s[st] = a;
    }

    // online softmax: C-layout rows live in lane halves -> 16-lane shfl reduce
    float alpha[8];
    #pragma unroll
    for (int r = 0; r < 8; ++r) {
      const int tg = t0 + hlf * 8 + r;
      float a0 = s[0][r] * scale;
      float a1 = s[1][r] * scale;
      if (s0 + n16 > tg)      a0 = -1e30f;     // causal mask
      if (s0 + 16 + n16 > tg) a1 = -1e30f;
      float rm = fmaxf(a0, a1);
      #pragma unroll
      for (int off = 1; off < 16; off <<= 1) rm = fmaxf(rm, __shfl_xor(rm, off, 32));
      const float mnew = fmaxf(mrun[r], rm);
      const float e0 = __expf(a0 - mnew);
      const float e1 = __expf(a1 - mnew);
      float rs = e0 + e1;
      #pragma unroll
      for (int off = 1; off < 16; off <<= 1) rs += __shfl_xor(rs, off, 32);
      const float al = __expf(mrun[r] - mnew);
      lrun[r] = lrun[r] * al + rs;
      mrun[r] = mnew;
      alpha[r] = al;
      s[0][r] = e0; s[1][r] = e1;
    }
    #pragma unroll
    for (int t = 0; t < 4; ++t)
      #pragma unroll
      for (int r = 0; r < 8; ++r) oacc[t][r] *= alpha[r];

    // stage P (C-layout -> row-major LDS -> A-layout reload), wave-private
    #pragma unroll
    for (int r = 0; r < 8; ++r) {
      plds[h][hlf * 8 + r][n16]      = (__bf16)s[0][r];
      plds[h][hlf * 8 + r][16 + n16] = (__bf16)s[1][r];
    }
    asm volatile("s_wait_dscnt 0" ::: "memory");
    v16bf pa = cat8(*(const v8bf*)&plds[h][n16][hlf * 8],
                    *(const v8bf*)&plds[h][n16][16 + hlf * 8]);

    // O += P V : 4 wmma over d tiles; B cols from transposed V (contiguous s)
    #pragma unroll
    for (int dt = 0; dt < 4; ++dt) {
      v16bf vb = cat8(*(const v8bf*)&vlds[dt * 16 + n16][hlf * 8],
                      *(const v8bf*)&vlds[dt * 16 + n16][16 + hlf * 8]);
      oacc[dt] = wmma_bf16(pa, vb, oacc[dt]);
    }
  }

  float inv[8];
  #pragma unroll
  for (int r = 0; r < 8; ++r) inv[r] = 1.0f / lrun[r];
  #pragma unroll
  for (int dt = 0; dt < 4; ++dt)
    #pragma unroll
    for (int r = 0; r < 8; ++r)
      Y[(btBase + t0 + hlf * 8 + r) * N_EMBD + h * HDIM + dt * 16 + n16] =
          (__bf16)(oacc[dt][r] * inv[r]);
}

// ============================================================
extern "C" void kernel_launch(void* const* d_in, const int* in_sizes, int n_in,
                              void* d_out, int out_size, void* d_ws, size_t ws_size,
                              hipStream_t stream) {
  const float* x   = (const float*)d_in[0];
  const float* Wd  = (const float*)d_in[1];
  const float* Wq  = (const float*)d_in[2];
  const float* Wkv = (const float*)d_in[3];
  const float* Wp  = (const float*)d_in[4];

  char* ws = (char*)d_ws;
  size_t off = 0;
  auto alloc = [&](size_t bytes) -> char* {
    char* p = ws + off;
    off += (bytes + 255) & ~(size_t)255;
    return p;
  };
  __bf16* xbf   = (__bf16*)alloc((size_t)BTOT * N_EMBD * 2);
  __bf16* latbf = (__bf16*)alloc((size_t)BTOT * LATENT * 2);
  __bf16* qbf   = (__bf16*)alloc((size_t)BTOT * N_EMBD * 2);
  __bf16* kbf   = (__bf16*)alloc((size_t)BTOT * HDIM * 2);
  __bf16* vtbf  = (__bf16*)alloc((size_t)BATCH * HDIM * TLEN * 2);
  __bf16* ybf   = (__bf16*)alloc((size_t)BTOT * N_EMBD * 2);
  __bf16* wdbf  = (__bf16*)alloc((size_t)N_EMBD * LATENT * 2);
  __bf16* wupbf = (__bf16*)alloc((size_t)LATENT * NUP * 2);
  __bf16* wpbf  = (__bf16*)alloc((size_t)N_EMBD * N_EMBD * 2);

  {
    size_t total = (size_t)BTOT * N_EMBD;
    prep_kernel<<<(int)((total + 255) / 256), 256, 0, stream>>>(
        x, Wd, Wq, Wkv, Wp, xbf, wdbf, wupbf, wpbf);
  }
  // latent = x @ Wc_down  (8192x1024 @ 1024x512) -> bf16
  wmma_gemm_kernel<<<dim3(BTOT / 256, LATENT / 64), 256, 0, stream>>>(
      xbf, wdbf, nullptr, latbf, nullptr, nullptr, BTOT, LATENT, N_EMBD, 0);
  // q|k|vT = latent @ [Wq_up|Wkv_up] (8192x512 @ 512x1152), split epilogue
  wmma_gemm_kernel<<<dim3(BTOT / 256, NUP / 64), 256, 0, stream>>>(
      latbf, wupbf, nullptr, qbf, kbf, vtbf, BTOT, NUP, LATENT, 2);
  // flash attention (k/v shared across the 16 heads of a workgroup)
  mla_attn_kernel<<<dim3(BATCH * (TLEN / 16)), 512, 0, stream>>>(qbf, kbf, vtbf, ybf);
  // out = y @ Wc_proj -> fp32 d_out
  wmma_gemm_kernel<<<dim3(BTOT / 256, N_EMBD / 64), 256, 0, stream>>>(
      ybf, wpbf, (float*)d_out, nullptr, nullptr, nullptr, BTOT, N_EMBD, N_EMBD, 1);

  (void)in_sizes; (void)n_in; (void)out_size; (void)ws_size;
}
